// SelfAttentionSeparate_50165218017586
// MI455X (gfx1250) — matrix operand order repaired and checked by reading, not compiled, not run on hardware
//
#include <hip/hip_runtime.h>

// Problem sizes
#define BB 2
#define SS 2048
#define EE 1024
#define HH 16
#define DD 64
#define RR 16

typedef __bf16 bf16;
typedef bf16 v16bf __attribute__((ext_vector_type(16)));
typedef bf16 v8bf  __attribute__((ext_vector_type(8)));
typedef float v8f  __attribute__((ext_vector_type(8)));

// ---------------------------------------------------------------------------
// WMMA helpers (CDNA5 gfx1250, wave32)
// ---------------------------------------------------------------------------
static __device__ __forceinline__ v16bf cat16(v8bf lo, v8bf hi) {
  return __builtin_shufflevector(lo, hi, 0,1,2,3,4,5,6,7,8,9,10,11,12,13,14,15);
}

// A operand: 16(M) x 32(K) bf16 from row-major src (row stride = ld elements).
// lanes 0-15: row M=lane, K 0-15; lanes 16-31: row M=lane-16, K 16-31.
static __device__ __forceinline__ v16bf load_A(const bf16* src, int ld, int lane) {
  const int half = lane >> 4, r = lane & 15;
  const bf16* p = src + (long)r * ld + half * 16;
  return cat16(*(const v8bf*)p, *(const v8bf*)(p + 8));
}

// B operand: 32(K) x 16(N) bf16 where element (k,n) = srcT[n*ld + k].
// lanes 0-15: col N=lane, K 0-15; lanes 16-31: col N=lane-16, K 16-31.
static __device__ __forceinline__ v16bf load_B(const bf16* srcT, int ld, int lane) {
  const int half = lane >> 4, n = lane & 15;
  const bf16* p = srcT + (long)n * ld + half * 16;
  return cat16(*(const v8bf*)p, *(const v8bf*)(p + 8));
}

static __device__ __forceinline__ v8f wmma_bf16(v16bf a, v16bf b, v8f c) {
  return __builtin_amdgcn_wmma_f32_16x16x32_bf16(false, a, false, b,
                                                 (short)0, c, false, false);
}

static __device__ __forceinline__ void lds_fence() {
  asm volatile("s_wait_dscnt 0" ::: "memory");
}

// LDS byte offset of a __shared__ object: low 32 bits of the generic address.
static __device__ __forceinline__ unsigned lds_offset(const void* p) {
  return (unsigned)(unsigned long long)p;
}

// ---------------------------------------------------------------------------
// CDNA5 async global->LDS copies (ASYNCcnt path) via inline asm
// ---------------------------------------------------------------------------
static __device__ __forceinline__ void async_load_b128(unsigned lds_off, const void* g) {
  asm volatile("global_load_async_to_lds_b128 %0, %1, off"
               :: "v"(lds_off), "v"(g) : "memory");
}
static __device__ __forceinline__ void async_wait0() {
  asm volatile("s_wait_asynccnt 0" ::: "memory");
}
static __device__ __forceinline__ void async_wait2() {   // allow 2 newest in flight
  asm volatile("s_wait_asynccnt 2" ::: "memory");
}

// ---------------------------------------------------------------------------
// Tensor Data Mover staging (TENSORcnt path) — clang-23 6-arg builtin:
// (uint32x4 g0, int32x8 g1, int32x4 g2, int32x4 g3, int32x8 gx, i32 cpol)
// ---------------------------------------------------------------------------
#if __has_builtin(__builtin_amdgcn_tensor_load_to_lds)
#define USE_TDM 1
typedef unsigned u32x4 __attribute__((ext_vector_type(4)));
typedef int i32x8 __attribute__((ext_vector_type(8)));
typedef int i32x4 __attribute__((ext_vector_type(4)));

// 2D tile load: `rows` rows of 32 bf16, row stride = row_stride elements.
// D# packed per CDNA5 ISA §8.3/§8.4 bit layout (data_size=2B, type=2).
static __device__ __forceinline__ void tdm_load_2d(const bf16* g, unsigned lds_off,
                                                   int rows, long row_stride) {
  unsigned long long ga = (unsigned long long)g;
  u32x4 g0;
  g0[0] = 1u;                                   // count=1, user descriptor
  g0[1] = lds_off;                              // lds_addr
  g0[2] = (unsigned)(ga & 0xFFFFFFFFu);         // global_addr[31:0]  (bits 95:64)
  g0[3] = (unsigned)((ga >> 32) & 0x1FFFFFFu)   // global_addr[56:32] (bits 120:96)
          | (2u << 30);                         // type=2 ("image")
  const unsigned td0 = 0x40000000u, td1 = 0x40000000u;  // huge dims: no OOB clip
  i32x8 g1;
  g1[0] = (int)(1u << 16);                      // data_size=1 (2 bytes), mask=0
  g1[1] = (int)((td0 & 0xFFFFu) << 16);         // tensor_dim0 lo16 (bits 63:48)
  g1[2] = (int)((td0 >> 16) | ((td1 & 0xFFFFu) << 16));   // td0 hi | td1 lo
  g1[3] = (int)(((td1 >> 16) & 0xFFFFu) | (32u << 16));   // td1 hi | tile_dim0=32
  g1[4] = (int)(unsigned)rows;                  // tile_dim1=rows, tile_dim2=0
  unsigned long long s0 = (unsigned long long)row_stride;
  g1[5] = (int)(s0 & 0xFFFFFFFFu);              // tensor_dim0_stride lo32
  g1[6] = (int)((s0 >> 32) & 0xFFFFu);          // stride0 hi16 | stride1 lo16=0
  g1[7] = 0;                                    // stride1 hi=0
  i32x4 z4 = {0, 0, 0, 0};
  i32x8 z8 = {0, 0, 0, 0, 0, 0, 0, 0};
  __builtin_amdgcn_tensor_load_to_lds(g0, g1, z4, z4, z8, 0);
}
#endif

// Cooperative fallback: rows x 32 bf16 tile, one b128 chunk per thread pass.
static __device__ __forceinline__ void stage_rows(const bf16* g, long ld, bf16* s,
                                                  int rows, int tid) {
  for (int c = tid; c < rows * 4; c += 256) {
    int r = c >> 2, qd = c & 3;
    *(v8bf*)(s + r * 32 + qd * 8) = *(const v8bf*)(g + (long)r * ld + qd * 8);
  }
}

// ---------------------------------------------------------------------------
// Prep kernels
// ---------------------------------------------------------------------------
__global__ void f32_to_bf16_kernel(const float* __restrict__ src,
                                   bf16* __restrict__ dst, long n) {
  long i = (long)blockIdx.x * blockDim.x + threadIdx.x;
  if (i < n) dst[i] = (bf16)src[i];
}

// src: (Hn, Rr, Cc) f32 -> dst: (Hn, Cc, Rr) bf16 (per-head transpose)
__global__ void transpose_to_bf16_kernel(const float* __restrict__ src,
                                         bf16* __restrict__ dst,
                                         int Hn, int Rr, int Cc) {
  long i = (long)blockIdx.x * blockDim.x + threadIdx.x;
  long n = (long)Hn * Rr * Cc;
  if (i >= n) return;
  long rr = i % Rr;
  long t  = i / Rr;
  long cc = t % Cc;
  long hh = t / Cc;
  dst[i] = (bf16)src[(hh * Rr + rr) * Cc + cc];
}

// ---------------------------------------------------------------------------
// Projection kernel: q = xWq+bq, k = xWk+bk, v = (xWvd+bvd)Wvu+bvu.
// 8 waves/block share (b,h); weight K-step tiles double-buffered in LDS and
// staged by the Tensor Data Mover (TDM issue for step i+1 overlaps the
// 9 WMMAs of step i). Each wave owns a 16-row s-tile.
// ---------------------------------------------------------------------------
__global__ void __launch_bounds__(256)
proj_kernel(const bf16* __restrict__ xb,    // (B,S,E)
            const bf16* __restrict__ WqT,   // (H,D,E)
            const bf16* __restrict__ WkT,   // (H,D,E)
            const bf16* __restrict__ WvdT,  // (H,R,E)
            const bf16* __restrict__ WvuT,  // (H,D,R)
            const float* __restrict__ bq, const float* __restrict__ bk,
            const float* __restrict__ bvd, const float* __restrict__ bvu,
            bf16* __restrict__ q,           // (B,H,S,D)
            bf16* __restrict__ k,           // (B,H,S,D)
            bf16* __restrict__ vT)          // (B,H,D,S)
{
  __shared__ __align__(16) bf16 wqs[2][64][32];
  __shared__ __align__(16) bf16 wks[2][64][32];
  __shared__ __align__(16) bf16 wvds[2][16][32];
  __shared__ __align__(16) bf16 vdstage[8][16][32];

  const int tid  = threadIdx.x;
  const int lane = tid & 31;
  const int w    = tid >> 5;
  const int gw0  = blockIdx.x * 8;          // 8 waves: same (b,h), consecutive stiles
  const int stile0 = gw0 & 127;             // S/16 = 128
  const int h      = (gw0 >> 7) & 15;
  const int b      = gw0 >> 11;
  const int sbase  = (stile0 + w) * 16;
  const int half   = lane >> 4, l16 = lane & 15;

  const bf16* xrow = xb + ((long)b * SS + sbase) * EE;
  const bf16* wqg  = WqT + (long)h * DD * EE;
  const bf16* wkg  = WkT + (long)h * DD * EE;
  const bf16* wvdg = WvdT + (long)h * RR * EE;

  const int NSTEP = EE / 32;  // 32 K-steps

  // prologue: stage step 0 into buffer 0
#if defined(USE_TDM)
  if (w == 0) {
    tdm_load_2d(wqg,  lds_offset(&wqs[0][0][0]),  64, EE);
    tdm_load_2d(wkg,  lds_offset(&wks[0][0][0]),  64, EE);
    tdm_load_2d(wvdg, lds_offset(&wvds[0][0][0]), 16, EE);
    __builtin_amdgcn_s_wait_tensorcnt((short)0);
  }
#else
  stage_rows(wqg,  EE, &wqs[0][0][0],  64, tid);
  stage_rows(wkg,  EE, &wks[0][0][0],  64, tid);
  stage_rows(wvdg, EE, &wvds[0][0][0], 16, tid);
#endif
  __syncthreads();

  v8f cq[4] = {}, ck[4] = {}, cvd = {};
  for (int i = 0; i < NSTEP; i++) {
    const int cur = i & 1, nxt = cur ^ 1;
    const int e0 = i * 32;
    // issue next step's DMA before computing this step
    if (i + 1 < NSTEP) {
#if defined(USE_TDM)
      if (w == 0) {
        tdm_load_2d(wqg + e0 + 32,  lds_offset(&wqs[nxt][0][0]),  64, EE);
        tdm_load_2d(wkg + e0 + 32,  lds_offset(&wks[nxt][0][0]),  64, EE);
        tdm_load_2d(wvdg + e0 + 32, lds_offset(&wvds[nxt][0][0]), 16, EE);
      }
#else
      stage_rows(wqg + e0 + 32,  EE, &wqs[nxt][0][0],  64, tid);
      stage_rows(wkg + e0 + 32,  EE, &wks[nxt][0][0],  64, tid);
      stage_rows(wvdg + e0 + 32, EE, &wvds[nxt][0][0], 16, tid);
#endif
    }

    v16bf a = load_A(xrow + e0, EE, lane);   // per-wave x rows (global, b128)
#pragma unroll
    for (int t = 0; t < 4; t++) {
      cq[t] = wmma_bf16(a, load_B(&wqs[cur][t * 16][0], 32, lane), cq[t]);
      ck[t] = wmma_bf16(a, load_B(&wks[cur][t * 16][0], 32, lane), ck[t]);
    }
    cvd = wmma_bf16(a, load_B(&wvds[cur][0][0], 32, lane), cvd);

#if defined(USE_TDM)
    if (w == 0)
      __builtin_amdgcn_s_wait_tensorcnt((short)0);  // next buffer landed
#endif
    __syncthreads();   // release next buffer / retire reads of current
  }

  // biases + store q/k (C layout: reg j -> row j + half*8, col l16)
  const long bhS = ((long)b * HH + h) * SS;
#pragma unroll
  for (int t = 0; t < 4; t++) {
    float bqv = bq[h * DD + t * 16 + l16];
    float bkv = bk[h * DD + t * 16 + l16];
#pragma unroll
    for (int j = 0; j < 8; j++) {
      int s = sbase + j + half * 8;
      long idx = (bhS + s) * DD + t * 16 + l16;
      q[idx] = (bf16)(cq[t][j] + bqv);
      k[idx] = (bf16)(ck[t][j] + bkv);
    }
  }

  // vd -> per-wave LDS (zero-padded 16x32 so the K=32 WMMA is valid)
  float bvdv = bvd[h * RR + l16];
#pragma unroll
  for (int j = 0; j < 8; j++) {
    int r = j + half * 8;
    vdstage[w][r][l16]      = (bf16)(cvd[j] + bvdv);
    vdstage[w][r][16 + l16] = (bf16)0.0f;
  }
  lds_fence();
  v16bf avd = load_A(&vdstage[w][0][0], 32, lane);

  const long bhD = ((long)b * HH + h) * DD;
#pragma unroll
  for (int t = 0; t < 4; t++) {
    v16bf bv = {};  // lanes 16-31 carry the zero K=16..31 padding
    if (lane < 16) {
      const bf16* p = WvuT + ((long)h * DD + t * 16 + l16) * RR;
      bv = cat16(*(const v8bf*)p, *(const v8bf*)(p + 8));
    }
    v8f cv = {};
    cv = wmma_bf16(avd, bv, cv);
    float bvuv = bvu[h * DD + t * 16 + l16];
#pragma unroll
    for (int j = 0; j < 8; j++) {
      int s = sbase + j + half * 8;
      long idx = (bhD + t * 16 + l16) * SS + s;  // (B,H,D,S)
      vT[idx] = (bf16)(cv[j] + bvuv);
    }
  }
}

// ---------------------------------------------------------------------------
// Flash-attention kernel (anti-causal: attend to t >= s).
// 8 waves/block share (b,h) and run a uniform key loop; K/V tiles are
// double-buffered in LDS via async global->LDS copies (issue for t+32
// overlaps compute of t). One wave owns one 16-query tile.
// ---------------------------------------------------------------------------
__global__ void __launch_bounds__(256)
attn_kernel(const bf16* __restrict__ q, const bf16* __restrict__ k,
            const bf16* __restrict__ vT, float* __restrict__ out)
{
  __shared__ __align__(16) bf16 ks[2][32][64];     // keys: t rows, d cols
  __shared__ __align__(16) bf16 vs[2][64][32];     // values: d rows, t cols
  __shared__ __align__(16) bf16 pstage[8][16][32]; // per-wave P staging

  const int tid  = threadIdx.x;
  const int lane = tid & 31;
  const int w    = tid >> 5;
  const int gw0  = blockIdx.x * 8;
  const int qtile0 = gw0 & 127;
  const int h      = (gw0 >> 7) & 15;
  const int b      = gw0 >> 11;
  const int qs     = (qtile0 + w) * 16;   // this wave's query tile
  const int qs0    = qtile0 * 16;         // block's first query (mult. of 128)
  const int half   = lane >> 4, l16 = lane & 15;
  const long bh    = (long)b * HH + h;

  const bf16* qp = q + (bh * SS + qs) * DD;
  const bf16* kp = k + bh * SS * DD;
  const bf16* vp = vT + bh * DD * SS;

  const int kr = tid >> 3, kc = tid & 7;   // K tile: 32 rows x 8 b128 chunks
  const int vr = tid >> 2, vc = tid & 3;   // V tile: 64 rows x 4 b128 chunks

  v16bf aq0 = load_A(qp, DD, lane);       // K = d 0..31
  v16bf aq1 = load_A(qp + 32, DD, lane);  // K = d 32..63

  float m[8], l[8];
  v8f acc[4] = {};
#pragma unroll
  for (int j = 0; j < 8; j++) { m[j] = -3e38f; l[j] = 0.0f; }

  const float scale = 0.03125f;  // 1/sqrt(E) = 1/sqrt(1024)

  // prologue: async-stage first K/V tile into buffer 0 (2 b128 per thread)
  async_load_b128(lds_offset(&ks[0][kr][kc * 8]),
                  kp + (long)(qs0 + kr) * DD + kc * 8);
  async_load_b128(lds_offset(&vs[0][vr][vc * 8]),
                  vp + (long)vr * SS + qs0 + vc * 8);

  for (int t = qs0; t < SS; t += 32) {
    const int cur = ((t - qs0) >> 5) & 1, nxt = cur ^ 1;
    // issue next tile's async copies, then drain only the older batch
    if (t + 32 < SS) {
      async_load_b128(lds_offset(&ks[nxt][kr][kc * 8]),
                      kp + (long)(t + 32 + kr) * DD + kc * 8);
      async_load_b128(lds_offset(&vs[nxt][vr][vc * 8]),
                      vp + (long)vr * SS + (t + 32) + vc * 8);
      async_wait2();   // in-order retirement: current tile's 2 chunks done
    } else {
      async_wait0();
    }
    __syncthreads();   // current tile visible to all waves
    if (t + 64 < SS)   // warm L2 one tile further ahead
      __builtin_prefetch((const void*)(kp + (long)(t + 64 + kr) * DD), 0, 1);

    if (t + 31 >= qs) {   // wave-uniform: this key tile matters for this wave
      // scores S = Q K^T : 2 N-tiles x 2 K-steps, operands from LDS
      v8f sc[2];
#pragma unroll
      for (int n = 0; n < 2; n++) {
        v8f c = {};
        c = wmma_bf16(aq0, load_B(&ks[cur][n * 16][0], 64, lane), c);
        c = wmma_bf16(aq1, load_B(&ks[cur][n * 16][32], 64, lane), c);
        sc[n] = c;
      }
      // scale + anti-causal mask (only tiles intersecting the diagonal)
#pragma unroll
      for (int n = 0; n < 2; n++) {
#pragma unroll
        for (int j = 0; j < 8; j++) {
          float f = sc[n][j] * scale;
          if (t < qs + 16) {
            int tg = t + n * 16 + l16;
            int sg = qs + j + half * 8;
            if (tg < sg) f = -3e38f;
          }
          sc[n][j] = f;
        }
      }
      // online softmax: row-reductions inside a 16-lane half
      float pr[2][8], alpha[8];
#pragma unroll
      for (int j = 0; j < 8; j++) {
        float rm = fmaxf(sc[0][j], sc[1][j]);
        rm = fmaxf(rm, __shfl_xor(rm, 1, 32));
        rm = fmaxf(rm, __shfl_xor(rm, 2, 32));
        rm = fmaxf(rm, __shfl_xor(rm, 4, 32));
        rm = fmaxf(rm, __shfl_xor(rm, 8, 32));
        float mn = fmaxf(m[j], rm);
        alpha[j] = __expf(m[j] - mn);
        m[j] = mn;
        float p0 = __expf(sc[0][j] - mn);
        float p1 = __expf(sc[1][j] - mn);
        pr[0][j] = p0; pr[1][j] = p1;
        float rs = p0 + p1;
        rs += __shfl_xor(rs, 1, 32);
        rs += __shfl_xor(rs, 2, 32);
        rs += __shfl_xor(rs, 4, 32);
        rs += __shfl_xor(rs, 8, 32);
        l[j] = l[j] * alpha[j] + rs;
      }
#pragma unroll
      for (int d = 0; d < 4; d++)
#pragma unroll
        for (int j = 0; j < 8; j++) acc[d][j] *= alpha[j];

      // P: C layout -> per-wave LDS -> A layout (16x32 bf16)
#pragma unroll
      for (int n = 0; n < 2; n++)
#pragma unroll
        for (int j = 0; j < 8; j++)
          pstage[w][j + half * 8][n * 16 + l16] = (bf16)pr[n][j];
      lds_fence();
      v16bf ap = load_A(&pstage[w][0][0], 32, lane);

      // acc += P V : V staged as (d rows, t cols) so B-operand is K-contiguous
#pragma unroll
      for (int d = 0; d < 4; d++)
        acc[d] = wmma_bf16(ap, load_B(&vs[cur][d * 16][0], 32, lane), acc[d]);
    }
    __syncthreads();   // retire reads of current buffer before it is reused
  }

  // epilogue: out[b,s,h*D+d] = acc / l
#pragma unroll
  for (int j = 0; j < 8; j++) l[j] = 1.0f / l[j];
#pragma unroll
  for (int d = 0; d < 4; d++) {
#pragma unroll
    for (int j = 0; j < 8; j++) {
      int s = qs + j + half * 8;
      long idx = ((long)b * SS + s) * (HH * DD) + h * DD + d * 16 + l16;
      out[idx] = acc[d][j] * l[j];
    }
  }
}

// ---------------------------------------------------------------------------
// Host launcher
// ---------------------------------------------------------------------------
extern "C" void kernel_launch(void* const* d_in, const int* in_sizes, int n_in,
                              void* d_out, int out_size, void* d_ws, size_t ws_size,
                              hipStream_t stream) {
  (void)in_sizes; (void)n_in; (void)out_size; (void)ws_size;
  const float* x   = (const float*)d_in[0];
  const float* Wq  = (const float*)d_in[1];
  const float* bq  = (const float*)d_in[2];
  const float* Wk  = (const float*)d_in[3];
  const float* bk  = (const float*)d_in[4];
  const float* Wvd = (const float*)d_in[5];
  const float* bvd = (const float*)d_in[6];
  const float* Wvu = (const float*)d_in[7];
  const float* bvu = (const float*)d_in[8];
  float* out = (float*)d_out;

  bf16* ws   = (bf16*)d_ws;
  bf16* xb   = ws;                          // B*S*E
  bf16* WqT  = xb   + (long)BB * SS * EE;   // H*D*E
  bf16* WkT  = WqT  + (long)HH * DD * EE;   // H*D*E
  bf16* WvdT = WkT  + (long)HH * DD * EE;   // H*R*E
  bf16* WvuT = WvdT + (long)HH * RR * EE;   // H*D*R
  bf16* qb   = WvuT + (long)HH * DD * RR;   // B*H*S*D
  bf16* kb   = qb   + (long)BB * HH * SS * DD;
  bf16* vTb  = kb   + (long)BB * HH * SS * DD;

  long nx = (long)BB * SS * EE;
  f32_to_bf16_kernel<<<(nx + 255) / 256, 256, 0, stream>>>(x, xb, nx);

  long nwq = (long)HH * EE * DD;
  transpose_to_bf16_kernel<<<(nwq + 255) / 256, 256, 0, stream>>>(Wq, WqT, HH, EE, DD);
  transpose_to_bf16_kernel<<<(nwq + 255) / 256, 256, 0, stream>>>(Wk, WkT, HH, EE, DD);
  long nwvd = (long)HH * EE * RR;
  transpose_to_bf16_kernel<<<(nwvd + 255) / 256, 256, 0, stream>>>(Wvd, WvdT, HH, EE, RR);
  long nwvu = (long)HH * RR * DD;
  transpose_to_bf16_kernel<<<(nwvu + 255) / 256, 256, 0, stream>>>(Wvu, WvuT, HH, RR, DD);

  // B*H*(S/16) waves / 8 waves per block = 512 blocks
  proj_kernel<<<512, 256, 0, stream>>>(xb, WqT, WkT, WvdT, WvuT,
                                       bq, bk, bvd, bvu, qb, kb, vTb);
  attn_kernel<<<512, 256, 0, stream>>>(qb, kb, vTb, out);
}
